// ProteinMPNN_59322088292938
// MI455X (gfx1250) — compile-verified, matrix operand order checked by reading, not run on hardware
//
#include <hip/hip_runtime.h>

#define L_N 8000
#define KNB 48
#define HD  128
#define FFN_DIM 512

typedef __bf16 v16bf __attribute__((ext_vector_type(16)));
typedef float  v8f   __attribute__((ext_vector_type(8)));

static __device__ inline v8f wmma_bf16(v16bf a, v16bf b, v8f c) {
    return __builtin_amdgcn_wmma_f32_16x16x32_bf16(false, a, false, b, (short)0, c, false, false);
}

// Load one 16x32 bf16 fragment (A-layout; B uses same layout on a pre-transposed
// matrix Bt[n][k]).  For lane l: row = l&15, and the 8 dwords are two contiguous
// 16-byte runs: k in [hi8, hi8+8) and [16+hi8, 16+hi8+8)  (hi8 = l>=16 ? 8 : 0).
// -> two b128 loads per fragment, landing directly in the fragment quads.
static __device__ inline v16bf load_frag(const __bf16* base, int stride) {
    const int lane = threadIdx.x & 31;
    const int r16  = lane & 15;
    const int hi8  = (lane >> 4) << 3;
    union { v16bf v; uint4 q[2]; } f;
    const __bf16* p0 = base + r16 * stride + hi8;
    f.q[0] = *(const uint4*)(p0);
    f.q[1] = *(const uint4*)(p0 + 16);
    return f.v;
}

static __device__ inline float gelu_f(float x) {
    return 0.5f * x * (1.0f + erff(x * 0.70710678118654752f));
}

static __device__ inline float wred_sum(float v) {
#pragma unroll
    for (int off = 16; off > 0; off >>= 1) v += __shfl_xor(v, off, 32);
    return v;
}

static __device__ inline uint4 pack8_bf16(float4 u, float4 v) {
    union { uint4 q; __bf16 h[8]; } pk;
    pk.h[0] = (__bf16)u.x; pk.h[1] = (__bf16)u.y;
    pk.h[2] = (__bf16)u.z; pk.h[3] = (__bf16)u.w;
    pk.h[4] = (__bf16)v.x; pk.h[5] = (__bf16)v.y;
    pk.h[6] = (__bf16)v.z; pk.h[7] = (__bf16)v.w;
    return pk.q;
}

// ---------------- weight packing: dst[n*K + k] = bf16(src[k*N + n]) ----------
__global__ void pack_wT(const float* __restrict__ src, __bf16* __restrict__ dst,
                        int K, int N) {
    int idx = blockIdx.x * blockDim.x + threadIdx.x;
    if (idx >= K * N) return;
    int n = idx / K, k = idx - n * K;
    dst[idx] = (__bf16)src[k * N + n];
}

// ---------------- Kernel 1: node message MLP + residual + LN1 ----------------
__launch_bounds__(256)
__global__ void node_msg(const float* __restrict__ hV, const float* __restrict__ hE,
                         const int* __restrict__ Eidx, const float* __restrict__ maskA,
                         const __bf16* __restrict__ W1t, const float* __restrict__ b1,
                         const __bf16* __restrict__ W2t, const float* __restrict__ b2,
                         const __bf16* __restrict__ W3t, const float* __restrict__ b3,
                         const float* __restrict__ n1s, const float* __restrict__ n1o,
                         float* __restrict__ hV1, __bf16* __restrict__ hV1b) {
    __shared__ __align__(16) __bf16 Asm[KNB * 384];   // 36 KB (reused as act2)
    __shared__ __align__(16) __bf16 act[KNB * HD];    // 12 KB
    __shared__ float  mask_sh[KNB];
    __shared__ float  dh_sh[HD];
    __shared__ float  red[8];

    const int i    = blockIdx.x;
    const int tid  = threadIdx.x;
    const int lane = tid & 31;
    const int w    = tid >> 5;
    const int r16  = lane & 15;
    const int rhi  = (lane >> 4) << 3;

    // ---- build A = [hV_i | hE_ik | hV_nbr], 8-elem chunks ----
    for (int p = tid; p < KNB * 48; p += 256) {
        int row = p / 48, ch = p - row * 48;
        int reg = ch >> 4;              // 0:self 1:edge 2:neighbor
        int c   = (ch & 15) << 3;       // elem offset within 128-col region
        const float* sp;
        if (reg == 0)      sp = hV + (size_t)i * HD + c;
        else if (reg == 1) sp = hE + ((size_t)i * KNB + row) * HD + c;
        else               sp = hV + (size_t)Eidx[i * KNB + row] * HD + c;
        float4 u = ((const float4*)sp)[0];
        float4 v = ((const float4*)sp)[1];
        *(uint4*)&Asm[row * 384 + (reg << 7) + c] = pack8_bf16(u, v);
    }
    if (tid < KNB) mask_sh[tid] = maskA[i * KNB + tid];
    __syncthreads();

    const int n0   = w * 16;
    const int colL = n0 + r16;

    // ---- GEMM1: (48x384) @ W1t -> gelu -> act ----
    v8f a0 = {}, a1 = {}, a2 = {};
    for (int ks = 0; ks < 12; ++ks) {
        int kb = ks * 32;
        v16bf bf = load_frag(W1t + (size_t)n0 * 384 + kb, 384);
        v16bf f0 = load_frag(&Asm[0 * 384 + kb], 384);
        v16bf f1 = load_frag(&Asm[16 * 384 + kb], 384);
        v16bf f2 = load_frag(&Asm[32 * 384 + kb], 384);
        a0 = wmma_bf16(f0, bf, a0);
        a1 = wmma_bf16(f1, bf, a1);
        a2 = wmma_bf16(f2, bf, a2);
    }
    {
        float bv = b1[colL];
#pragma unroll
        for (int j = 0; j < 8; ++j) {
            act[(0  + j + rhi) * HD + colL] = (__bf16)gelu_f(a0[j] + bv);
            act[(16 + j + rhi) * HD + colL] = (__bf16)gelu_f(a1[j] + bv);
            act[(32 + j + rhi) * HD + colL] = (__bf16)gelu_f(a2[j] + bv);
        }
    }
    __syncthreads();

    // ---- GEMM2: act @ W2t -> gelu -> act2 (reuse Asm) ----
    __bf16* act2 = Asm;
    a0 = v8f{}; a1 = v8f{}; a2 = v8f{};
    for (int ks = 0; ks < 4; ++ks) {
        int kb = ks * 32;
        v16bf bf = load_frag(W2t + (size_t)n0 * HD + kb, HD);
        v16bf f0 = load_frag(&act[0 * HD + kb], HD);
        v16bf f1 = load_frag(&act[16 * HD + kb], HD);
        v16bf f2 = load_frag(&act[32 * HD + kb], HD);
        a0 = wmma_bf16(f0, bf, a0);
        a1 = wmma_bf16(f1, bf, a1);
        a2 = wmma_bf16(f2, bf, a2);
    }
    {
        float bv = b2[colL];
#pragma unroll
        for (int j = 0; j < 8; ++j) {
            act2[(0  + j + rhi) * HD + colL] = (__bf16)gelu_f(a0[j] + bv);
            act2[(16 + j + rhi) * HD + colL] = (__bf16)gelu_f(a1[j] + bv);
            act2[(32 + j + rhi) * HD + colL] = (__bf16)gelu_f(a2[j] + bv);
        }
    }
    __syncthreads();

    // ---- GEMM3: act2 @ W3t -> masked row-sum ----
    a0 = v8f{}; a1 = v8f{}; a2 = v8f{};
    for (int ks = 0; ks < 4; ++ks) {
        int kb = ks * 32;
        v16bf bf = load_frag(W3t + (size_t)n0 * HD + kb, HD);
        v16bf f0 = load_frag(&act2[0 * HD + kb], HD);
        v16bf f1 = load_frag(&act2[16 * HD + kb], HD);
        v16bf f2 = load_frag(&act2[32 * HD + kb], HD);
        a0 = wmma_bf16(f0, bf, a0);
        a1 = wmma_bf16(f1, bf, a1);
        a2 = wmma_bf16(f2, bf, a2);
    }
    {
        float bv = b3[colL];
        float part = 0.f;
#pragma unroll
        for (int j = 0; j < 8; ++j) {
            part += (a0[j] + bv) * mask_sh[0  + j + rhi];
            part += (a1[j] + bv) * mask_sh[16 + j + rhi];
            part += (a2[j] + bv) * mask_sh[32 + j + rhi];
        }
        part += __shfl_xor(part, 16, 32);
        if (lane < 16) dh_sh[colL] = part * (1.0f / 30.0f);
    }
    __syncthreads();

    // ---- residual + LN1 ----
    float x = 0.f;
    if (tid < HD) x = hV[(size_t)i * HD + tid] + dh_sh[tid];
    float s = wred_sum(x);
    float q = wred_sum(x * x);
    if (lane == 0 && w < 4) { red[w * 2] = s; red[w * 2 + 1] = q; }
    __syncthreads();
    if (tid < HD) {
        float sum = red[0] + red[2] + red[4] + red[6];
        float sq  = red[1] + red[3] + red[5] + red[7];
        float mu  = sum * (1.0f / HD);
        float var = sq * (1.0f / HD) - mu * mu;
        float rs  = rsqrtf(var + 1e-5f);
        float y   = (x - mu) * rs * n1s[tid] + n1o[tid];
        hV1 [(size_t)i * HD + tid] = y;
        hV1b[(size_t)i * HD + tid] = (__bf16)y;
    }
}

// ---------------- Kernel 2: FFN + residual + LN2 + mask ----------------------
__launch_bounds__(256)
__global__ void ffn_ln(const float* __restrict__ hV1, const __bf16* __restrict__ hV1b,
                       const __bf16* __restrict__ Wit, const float* __restrict__ bi,
                       const __bf16* __restrict__ Wot, const float* __restrict__ bo,
                       const float* __restrict__ n2s, const float* __restrict__ n2o,
                       const float* __restrict__ maskV,
                       float* __restrict__ hVout, __bf16* __restrict__ hV2b) {
    __shared__ __align__(16) __bf16 Abuf[32 * HD];        // 8 KB
    __shared__ __align__(16) __bf16 act[32 * FFN_DIM];    // 32 KB
    __shared__ __align__(16) float  dsh[32 * HD];         // 16 KB

    const int node0 = blockIdx.x * 32;
    const int tid   = threadIdx.x;
    const int lane  = tid & 31;
    const int w     = tid >> 5;
    const int r16   = lane & 15;
    const int rhi   = (lane >> 4) << 3;

    {
        const uint4* src = (const uint4*)(hV1b + (size_t)node0 * HD);
        uint4* dst = (uint4*)Abuf;
        for (int p = tid; p < 32 * HD / 8; p += 256) dst[p] = src[p];
    }
    __syncthreads();

    // ---- GEMM1: (32x128) @ Wit -> gelu -> act (32x512) ----
    v8f acc[4][2] = {};
    for (int ks = 0; ks < 4; ++ks) {
        int kb = ks * 32;
        v16bf f0 = load_frag(&Abuf[0 * HD + kb], HD);
        v16bf f1 = load_frag(&Abuf[16 * HD + kb], HD);
#pragma unroll
        for (int nt = 0; nt < 4; ++nt) {
            int ncol = (w + nt * 8) * 16;
            v16bf bf = load_frag(Wit + (size_t)ncol * HD + kb, HD);
            acc[nt][0] = wmma_bf16(f0, bf, acc[nt][0]);
            acc[nt][1] = wmma_bf16(f1, bf, acc[nt][1]);
        }
    }
#pragma unroll
    for (int nt = 0; nt < 4; ++nt) {
        int col = (w + nt * 8) * 16 + r16;
        float bv = bi[col];
#pragma unroll
        for (int j = 0; j < 8; ++j) {
            act[(0  + j + rhi) * FFN_DIM + col] = (__bf16)gelu_f(acc[nt][0][j] + bv);
            act[(16 + j + rhi) * FFN_DIM + col] = (__bf16)gelu_f(acc[nt][1][j] + bv);
        }
    }
    __syncthreads();

    // ---- GEMM2: act @ Wot ----
    v8f o0 = {}, o1 = {};
    const int n0 = w * 16;
    for (int ks = 0; ks < 16; ++ks) {
        int kb = ks * 32;
        v16bf bf = load_frag(Wot + (size_t)n0 * FFN_DIM + kb, FFN_DIM);
        v16bf f0 = load_frag(&act[0 * FFN_DIM + kb], FFN_DIM);
        v16bf f1 = load_frag(&act[16 * FFN_DIM + kb], FFN_DIM);
        o0 = wmma_bf16(f0, bf, o0);
        o1 = wmma_bf16(f1, bf, o1);
    }
    {
        float bv = bo[n0 + r16];
#pragma unroll
        for (int j = 0; j < 8; ++j) {
            dsh[(0  + j + rhi) * HD + n0 + r16] = o0[j] + bv;
            dsh[(16 + j + rhi) * HD + n0 + r16] = o1[j] + bv;
        }
    }
    __syncthreads();

    // ---- residual + LN2 + mask: wave w handles rows 4w..4w+3 ----
#pragma unroll
    for (int rr = 0; rr < 4; ++rr) {
        int r = w * 4 + rr;
        size_t gbase = (size_t)(node0 + r) * HD;
        float xv[4], s = 0.f, q = 0.f;
#pragma unroll
        for (int qq = 0; qq < 4; ++qq) {
            int c = lane + 32 * qq;
            float x = hV1[gbase + c] + dsh[r * HD + c];
            xv[qq] = x; s += x; q += x * x;
        }
        s = wred_sum(s); q = wred_sum(q);
        float mu  = s * (1.0f / HD);
        float var = q * (1.0f / HD) - mu * mu;
        float rs  = rsqrtf(var + 1e-5f);
        float mk  = maskV[node0 + r];
#pragma unroll
        for (int qq = 0; qq < 4; ++qq) {
            int c = lane + 32 * qq;
            float y = ((xv[qq] - mu) * rs * n2s[c] + n2o[c]) * mk;
            hVout[gbase + c] = y;
            hV2b [gbase + c] = (__bf16)y;
        }
    }
}

// ---------------- Kernel 3: edge-update MLP + residual + LN3 ----------------
__launch_bounds__(256)
__global__ void edge_upd(const float* __restrict__ hE, const int* __restrict__ Eidx,
                         const __bf16* __restrict__ hV2b,
                         const __bf16* __restrict__ W11t, const float* __restrict__ b11,
                         const __bf16* __restrict__ W12t, const float* __restrict__ b12,
                         const __bf16* __restrict__ W13t, const float* __restrict__ b13,
                         const float* __restrict__ n3s, const float* __restrict__ n3o,
                         float* __restrict__ hEout) {
    __shared__ __align__(16) __bf16 Asm[KNB * 384];   // 36 KB (reused as act2)
    __shared__ __align__(16) __bf16 act[KNB * HD];    // 12 KB
    __shared__ __align__(16) float  msh[KNB * HD];    // 24 KB

    const int i    = blockIdx.x;
    const int tid  = threadIdx.x;
    const int lane = tid & 31;
    const int w    = tid >> 5;
    const int r16  = lane & 15;
    const int rhi  = (lane >> 4) << 3;

    for (int p = tid; p < KNB * 48; p += 256) {
        int row = p / 48, ch = p - row * 48;
        int reg = ch >> 4;
        int c   = (ch & 15) << 3;
        uint4 val;
        if (reg == 1) {
            const float* sp = hE + ((size_t)i * KNB + row) * HD + c;
            val = pack8_bf16(((const float4*)sp)[0], ((const float4*)sp)[1]);
        } else {
            size_t node = (reg == 0) ? (size_t)i : (size_t)Eidx[i * KNB + row];
            val = ((const uint4*)(hV2b + node * HD))[c >> 3];
        }
        *(uint4*)&Asm[row * 384 + (reg << 7) + c] = val;
    }
    __syncthreads();

    const int n0   = w * 16;
    const int colL = n0 + r16;

    v8f a0 = {}, a1 = {}, a2 = {};
    for (int ks = 0; ks < 12; ++ks) {
        int kb = ks * 32;
        v16bf bf = load_frag(W11t + (size_t)n0 * 384 + kb, 384);
        v16bf f0 = load_frag(&Asm[0 * 384 + kb], 384);
        v16bf f1 = load_frag(&Asm[16 * 384 + kb], 384);
        v16bf f2 = load_frag(&Asm[32 * 384 + kb], 384);
        a0 = wmma_bf16(f0, bf, a0);
        a1 = wmma_bf16(f1, bf, a1);
        a2 = wmma_bf16(f2, bf, a2);
    }
    {
        float bv = b11[colL];
#pragma unroll
        for (int j = 0; j < 8; ++j) {
            act[(0  + j + rhi) * HD + colL] = (__bf16)gelu_f(a0[j] + bv);
            act[(16 + j + rhi) * HD + colL] = (__bf16)gelu_f(a1[j] + bv);
            act[(32 + j + rhi) * HD + colL] = (__bf16)gelu_f(a2[j] + bv);
        }
    }
    __syncthreads();

    __bf16* act2 = Asm;
    a0 = v8f{}; a1 = v8f{}; a2 = v8f{};
    for (int ks = 0; ks < 4; ++ks) {
        int kb = ks * 32;
        v16bf bf = load_frag(W12t + (size_t)n0 * HD + kb, HD);
        v16bf f0 = load_frag(&act[0 * HD + kb], HD);
        v16bf f1 = load_frag(&act[16 * HD + kb], HD);
        v16bf f2 = load_frag(&act[32 * HD + kb], HD);
        a0 = wmma_bf16(f0, bf, a0);
        a1 = wmma_bf16(f1, bf, a1);
        a2 = wmma_bf16(f2, bf, a2);
    }
    {
        float bv = b12[colL];
#pragma unroll
        for (int j = 0; j < 8; ++j) {
            act2[(0  + j + rhi) * HD + colL] = (__bf16)gelu_f(a0[j] + bv);
            act2[(16 + j + rhi) * HD + colL] = (__bf16)gelu_f(a1[j] + bv);
            act2[(32 + j + rhi) * HD + colL] = (__bf16)gelu_f(a2[j] + bv);
        }
    }
    __syncthreads();

    a0 = v8f{}; a1 = v8f{}; a2 = v8f{};
    for (int ks = 0; ks < 4; ++ks) {
        int kb = ks * 32;
        v16bf bf = load_frag(W13t + (size_t)n0 * HD + kb, HD);
        v16bf f0 = load_frag(&act2[0 * HD + kb], HD);
        v16bf f1 = load_frag(&act2[16 * HD + kb], HD);
        v16bf f2 = load_frag(&act2[32 * HD + kb], HD);
        a0 = wmma_bf16(f0, bf, a0);
        a1 = wmma_bf16(f1, bf, a1);
        a2 = wmma_bf16(f2, bf, a2);
    }
    {
        float bv = b13[colL];
#pragma unroll
        for (int j = 0; j < 8; ++j) {
            msh[(0  + j + rhi) * HD + colL] = a0[j] + bv;
            msh[(16 + j + rhi) * HD + colL] = a1[j] + bv;
            msh[(32 + j + rhi) * HD + colL] = a2[j] + bv;
        }
    }
    __syncthreads();

    // ---- residual + LN3: wave w handles rows 6w..6w+5 ----
#pragma unroll
    for (int rr = 0; rr < 6; ++rr) {
        int r = w * 6 + rr;
        size_t gbase = ((size_t)i * KNB + r) * HD;
        float xv[4], s = 0.f, q = 0.f;
#pragma unroll
        for (int qq = 0; qq < 4; ++qq) {
            int c = lane + 32 * qq;
            float x = hE[gbase + c] + msh[r * HD + c];
            xv[qq] = x; s += x; q += x * x;
        }
        s = wred_sum(s); q = wred_sum(q);
        float mu  = s * (1.0f / HD);
        float var = q * (1.0f / HD) - mu * mu;
        float rs  = rsqrtf(var + 1e-5f);
#pragma unroll
        for (int qq = 0; qq < 4; ++qq) {
            int c = lane + 32 * qq;
            hEout[gbase + c] = (xv[qq] - mu) * rs * n3s[c] + n3o[c];
        }
    }
}

extern "C" void kernel_launch(void* const* d_in, const int* in_sizes, int n_in,
                              void* d_out, int out_size, void* d_ws, size_t ws_size,
                              hipStream_t stream) {
    const float* hV    = (const float*)d_in[0];
    const float* hE    = (const float*)d_in[1];
    const int*   Eidx  = (const int*)d_in[2];
    const float* maskV = (const float*)d_in[3];
    const float* maskA = (const float*)d_in[4];
    const float* W1 = (const float*)d_in[5],  *b1  = (const float*)d_in[6];
    const float* W2 = (const float*)d_in[7],  *b2  = (const float*)d_in[8];
    const float* W3 = (const float*)d_in[9],  *b3  = (const float*)d_in[10];
    const float* W11 = (const float*)d_in[11], *b11 = (const float*)d_in[12];
    const float* W12 = (const float*)d_in[13], *b12 = (const float*)d_in[14];
    const float* W13 = (const float*)d_in[15], *b13 = (const float*)d_in[16];
    const float* Wi = (const float*)d_in[17], *bi  = (const float*)d_in[18];
    const float* Wo = (const float*)d_in[19], *bo  = (const float*)d_in[20];
    const float* n1s = (const float*)d_in[21], *n1o = (const float*)d_in[22];
    const float* n2s = (const float*)d_in[23], *n2o = (const float*)d_in[24];
    const float* n3s = (const float*)d_in[25], *n3o = (const float*)d_in[26];

    char* ws = (char*)d_ws;
    size_t off = 0;
    auto alloc = [&](size_t bytes) -> void* {
        void* p = ws + off;
        off += (bytes + 255) & ~(size_t)255;
        return p;
    };
    __bf16* W1t  = (__bf16*)alloc((size_t)384 * HD * 2);
    __bf16* W2t  = (__bf16*)alloc((size_t)HD * HD * 2);
    __bf16* W3t  = (__bf16*)alloc((size_t)HD * HD * 2);
    __bf16* W11t = (__bf16*)alloc((size_t)384 * HD * 2);
    __bf16* W12t = (__bf16*)alloc((size_t)HD * HD * 2);
    __bf16* W13t = (__bf16*)alloc((size_t)HD * HD * 2);
    __bf16* Wit  = (__bf16*)alloc((size_t)HD * FFN_DIM * 2);
    __bf16* Wot  = (__bf16*)alloc((size_t)FFN_DIM * HD * 2);
    float*  hV1  = (float*)alloc((size_t)L_N * HD * 4);
    __bf16* hV1b = (__bf16*)alloc((size_t)L_N * HD * 2);
    __bf16* hV2b = (__bf16*)alloc((size_t)L_N * HD * 2);

    auto pack = [&](const float* s, __bf16* d, int K, int N) {
        int tot = K * N;
        pack_wT<<<(tot + 255) / 256, 256, 0, stream>>>(s, d, K, N);
    };
    pack(W1, W1t, 384, HD);
    pack(W2, W2t, HD, HD);
    pack(W3, W3t, HD, HD);
    pack(W11, W11t, 384, HD);
    pack(W12, W12t, HD, HD);
    pack(W13, W13t, HD, HD);
    pack(Wi, Wit, HD, FFN_DIM);
    pack(Wo, Wot, FFN_DIM, HD);

    float* hVout = (float*)d_out;
    float* hEout = (float*)d_out + (size_t)L_N * HD;

    node_msg<<<L_N, 256, 0, stream>>>(hV, hE, Eidx, maskA,
                                      W1t, b1, W2t, b2, W3t, b3,
                                      n1s, n1o, hV1, hV1b);
    ffn_ln<<<L_N / 32, 256, 0, stream>>>(hV1, hV1b, Wit, bi, Wot, bo,
                                         n2s, n2o, maskV, hVout, hV2b);
    edge_upd<<<L_N, 256, 0, stream>>>(hE, Eidx, hV2b,
                                      W11t, b11, W12t, b12, W13t, b13,
                                      n3s, n3o, hEout);
    (void)in_sizes; (void)n_in; (void)out_size; (void)ws_size;
}
